// CaloClusterNet_4595615007038
// MI455X (gfx1250) — matrix operand order repaired and checked by reading, not compiled, not run on hardware
//
#include <hip/hip_runtime.h>
#include <hip/hip_bf16.h>

typedef _Float16 f16;
typedef __attribute__((ext_vector_type(16))) _Float16 v16h;
typedef __attribute__((ext_vector_type(8)))  _Float16 v8h;
typedef __attribute__((ext_vector_type(8)))  float    v8f;

#define LDA 40   // f16 columns in staged A chunk (32 used); 80 B row stride (16B aligned)
#define LDT 104  // f16 columns in staged tmp (96 used); 208 B row stride (16B aligned)

// ---------------------------------------------------------------------------
// Weight conversion: fp32 [K, Nsrc] -> f16 fragment-ordered buffer.
// Layout: [kc][nt][lane][i] with k = kc*32 + ((lane&16)?16:0) + i,
//                            n = nt*16 + (lane&15). Zero padded.
// ---------------------------------------------------------------------------
__global__ void cvt_w_frag_kernel(const float* __restrict__ src, f16* __restrict__ dst,
                                  int K, int Nsrc, int Kpad) {
    int i = blockIdx.x * blockDim.x + threadIdx.x;
    int total = Kpad * 96;
    if (i >= total) return;
    int ii   = i & 15;
    int lane = (i >> 4) & 31;
    int nt   = (i >> 9) % 6;
    int kc   = i / (512 * 6);
    int k = kc * 32 + ((lane & 16) ? 16 : 0) + ii;
    int n = nt * 16 + (lane & 15);
    float v = (k < K && n < Nsrc) ? src[k * Nsrc + n] : 0.0f;
    dst[i] = (f16)v;
}

// ---------------------------------------------------------------------------
// segment_sum: agg[dst[e], :] += e[e, :]   (float4 + fp32 atomics)
// ---------------------------------------------------------------------------
__global__ void seg_sum_kernel(const float* __restrict__ e,
                               const long long* __restrict__ dst,
                               float* __restrict__ agg, int E) {
    int i = blockIdx.x * blockDim.x + threadIdx.x;
    int total = E * 24;
    if (i >= total) return;
    int edge = i / 24;
    int c4 = (i % 24) * 4;
    long long d = dst[edge];
    const float4 v = *(const float4*)(e + (long long)edge * 96 + c4);
    float* a = agg + d * 96 + c4;
    atomicAdd(a + 0, v.x);
    atomicAdd(a + 1, v.y);
    atomicAdd(a + 2, v.z);
    atomicAdd(a + 3, v.w);
}

// ---------------------------------------------------------------------------
// WMMA fragment helpers (contiguous k = 16*(lane/16) + i order in A and B, wave32)
// ---------------------------------------------------------------------------
__device__ __forceinline__ v16h load_afrag(const f16* base, int ld, int lane) {
    int m  = lane & 15;
    int kb = (lane & 16) ? 16 : 0;
    const f16* r = base + m * ld + kb;
    v8h lo = *(const v8h*)(r);
    v8h hi = *(const v8h*)(r + 8);
    return __builtin_shufflevector(lo, hi, 0, 1, 2, 3, 4, 5, 6, 7,
                                           8, 9, 10, 11, 12, 13, 14, 15);
}

__device__ __forceinline__ v16h load_bfrag_f(const f16* fbase, int nt, int lane) {
    const v8h* p = (const v8h*)(fbase + ((nt * 32 + lane) << 4));
    v8h lo = p[0];
    v8h hi = p[1];
    return __builtin_shufflevector(lo, hi, 0, 1, 2, 3, 4, 5, 6, 7,
                                           8, 9, 10, 11, 12, 13, 14, 15);
}

__device__ __forceinline__ float gelu_exact(float x) {
    return 0.5f * x * (1.0f + erff(x * 0.70710678118654752f));
}

// ---------------------------------------------------------------------------
// Fused MLP2:  out = gelu(A @ W1 + b1) @ W2 + b2 (+ resid)
//   mode 0: A = X fp32 [rows, Kin]                      (encoders, Kin = 6/8)
//   mode 1: A = [U16 | V32]  f16 mirror + fp32 agg      (Kin = 192)
//   mode 2: A = [U16[idxS] | U16[idxD] | V16] all f16   (Kin = 288)
//   mode 3: A = U16 f16 [rows,96]                       (node head)
// Writes fp32 out (ldo, Hout cols) and optional f16 mirror out16 [rows,96].
// 256 threads = 8 waves; 128 rows per block; each wave owns 16 rows.
// ---------------------------------------------------------------------------
__global__ __launch_bounds__(256)
void fused_mlp_kernel(int mode, int rows, int Kin,
                      const float* __restrict__ X,
                      const f16* __restrict__ U16,
                      const f16* __restrict__ V16,
                      const float* __restrict__ V32,
                      const long long* __restrict__ idxS,
                      const long long* __restrict__ idxD,
                      const f16* __restrict__ W1, const float* __restrict__ b1,
                      const f16* __restrict__ W2, const float* __restrict__ b2,
                      float* __restrict__ out, int ldo,
                      const float* __restrict__ resid, int Hout,
                      f16* __restrict__ out16) {
    __shared__ __align__(16) f16 sA[128 * LDA];        // staged A chunk       (10240 B)
    __shared__ __align__(16) f16 sW1[6 * 32 * 16];     // W1 chunk frag-order  ( 6144 B)
    __shared__ __align__(16) f16 sTmp[128 * LDT];      // hidden / out-stage   (26624 B)
    __shared__ __align__(16) f16 sW2[3 * 6 * 32 * 16]; // W2 frag-order        (18432 B)

    const int tid  = threadIdx.x;
    const int wv   = tid >> 5;
    const int lane = tid & 31;
    const int row0 = blockIdx.x * 128;

    // stage full W2 once (first k-loop barrier pair makes it visible)
    {
        const uint4* w2u = (const uint4*)W2;   // 9216 f16 = 1152 uint4
        for (int t = tid; t < 1152; t += 256) ((uint4*)sW2)[t] = w2u[t];
    }

    v8f acc[6];
    const v8f vzero = {0.f, 0.f, 0.f, 0.f, 0.f, 0.f, 0.f, 0.f};
#pragma unroll
    for (int j = 0; j < 6; ++j) acc[j] = vzero;

    const int Kpad = (Kin + 31) & ~31;
    const int nK1  = Kpad >> 5;

    // ---------------- GEMM1: A[128,Kpad] x W1[Kpad,96] ----------------
    for (int kc = 0; kc < nK1; ++kc) {
        const int k0 = kc << 5;
        const int region = k0 / 96;
        const int off    = k0 % 96;
        __syncthreads();
        // ---- stage A chunk ----
        if (mode == 0) {
            // small fp32 input, scalar convert
            for (int t = tid; t < 128 * 32; t += 256) {
                int r = t >> 5, k = t & 31;
                int row = row0 + r;
                float v = 0.0f;
                int kk = k0 + k;
                if (row < rows && kk < Kin) v = X[(long long)row * Kin + kk];
                sA[r * LDA + k] = (f16)v;
            }
        } else if (mode == 1 && region == 1) {
            // agg fp32 -> f16 convert path
            for (int t = tid; t < 128 * 8; t += 256) {
                int r = t >> 3, q = t & 7;
                int row = row0 + r;
                bool ok = row < rows;
                int rc = ok ? row : 0;
                const float* base = V32 + (long long)rc * 96 + off;
                float4 v = ok ? *(const float4*)(base + q * 4)
                              : make_float4(0.f, 0.f, 0.f, 0.f);
                union { f16 h[4]; uint2 u; } pk;
                pk.h[0] = (f16)v.x; pk.h[1] = (f16)v.y;
                pk.h[2] = (f16)v.z; pk.h[3] = (f16)v.w;
                *(uint2*)(sA + r * LDA + q * 4) = pk.u;
            }
        } else {
            // f16 mirror direct copy (16B per thread-iteration)
            for (int t = tid; t < 128 * 4; t += 256) {
                int r = t >> 2, q = t & 3;
                int row = row0 + r;
                bool ok = row < rows;
                int rc = ok ? row : 0;
                const f16* base;
                if (mode == 2) {
                    if (region == 0)      base = U16 + (long long)idxS[rc] * 96 + off;
                    else if (region == 1) base = U16 + (long long)idxD[rc] * 96 + off;
                    else                  base = V16 + (long long)rc * 96 + off;
                } else {  // mode 1 region 0, mode 3
                    base = U16 + (long long)rc * 96 + off;
                }
                uint4 v = ok ? *(const uint4*)(base + q * 8)
                             : make_uint4(0u, 0u, 0u, 0u);
                *(uint4*)(sA + r * LDA + q * 8) = v;
            }
        }
        // ---- stage W1 chunk (frag-ordered, contiguous) ----
        {
            const uint4* w1u = (const uint4*)(W1 + kc * 3072);
            for (int t = tid; t < 384; t += 256) ((uint4*)sW1)[t] = w1u[t];
        }
        __syncthreads();

        // preload all B fragments, then burst the WMMAs
        v16h bf[6];
#pragma unroll
        for (int nt = 0; nt < 6; ++nt) bf[nt] = load_bfrag_f(sW1, nt, lane);
        const v16h a = load_afrag(sA + (wv * 16) * LDA, LDA, lane);
#pragma unroll
        for (int nt = 0; nt < 6; ++nt)
            acc[nt] = __builtin_amdgcn_wmma_f32_16x16x32_f16(
                false, a, false, bf[nt], (short)0, acc[nt], false, false);
    }

    // bias + GELU -> f16 LDS tmp (each wave writes only its own 16 rows)
    {
        const int n  = lane & 15;
        const int mb = (lane & 16) ? 8 : 0;
#pragma unroll
        for (int nt = 0; nt < 6; ++nt) {
            const int c = nt * 16 + n;
            const float bb = b1[c];
#pragma unroll
            for (int i = 0; i < 8; ++i) {
                int rl = wv * 16 + mb + i;
                float vv = gelu_exact(acc[nt][i] + bb);
                sTmp[rl * LDT + c] = (f16)vv;
            }
        }
    }

    // ---------------- GEMM2: tmp[128,96] x W2[96,96] ----------------
    const int NT2 = (Hout == 96) ? 6 : 1;
#pragma unroll
    for (int j = 0; j < 6; ++j) acc[j] = vzero;

#pragma unroll
    for (int kc2 = 0; kc2 < 3; ++kc2) {
        const v16h a = load_afrag(sTmp + (wv * 16) * LDT + kc2 * 32, LDT, lane);
        v16h bf[6];
        for (int nt = 0; nt < NT2; ++nt)
            bf[nt] = load_bfrag_f(sW2 + kc2 * 3072, nt, lane);
        for (int nt = 0; nt < NT2; ++nt)
            acc[nt] = __builtin_amdgcn_wmma_f32_16x16x32_f16(
                false, a, false, bf[nt], (short)0, acc[nt], false, false);
    }

    // bias (+ residual); store fp32 master, stash f16 mirror rows in sTmp
    {
        const int n  = lane & 15;
        const int mb = (lane & 16) ? 8 : 0;
        for (int nt = 0; nt < NT2; ++nt) {
            const int c = nt * 16 + n;
            const float bb2 = b2[c];
#pragma unroll
            for (int i = 0; i < 8; ++i) {
                int rl  = wv * 16 + mb + i;
                int row = row0 + rl;
                float vv = acc[nt][i] + bb2;
                if (row < rows && c < Hout) {
                    long long o = (long long)row * ldo + c;
                    if (resid) vv += resid[o];
                    out[o] = vv;
                }
                if (out16) sTmp[rl * LDT + c] = (f16)vv;
            }
        }
    }

    // coalesced f16 mirror write (192 B per row via uint4)
    if (out16) {
        __syncthreads();
        for (int t = tid; t < 128 * 12; t += 256) {
            int r = t / 12, q = t % 12;
            int row = row0 + r;
            if (row < rows)
                *(uint4*)(out16 + (long long)row * 96 + q * 8) =
                    *(const uint4*)(sTmp + r * LDT + q * 8);
        }
    }
}

// ---------------------------------------------------------------------------
// Host side
// ---------------------------------------------------------------------------
extern "C" void kernel_launch(void* const* d_in, const int* in_sizes, int n_in,
                              void* d_out, int out_size, void* d_ws, size_t ws_size,
                              hipStream_t stream) {
    const float*     x    = (const float*)d_in[0];
    const long long* ei   = (const long long*)d_in[1];
    const float*     ea   = (const float*)d_in[2];
    const float* ne_w1 = (const float*)d_in[3],  *ne_b1 = (const float*)d_in[4];
    const float* ne_w2 = (const float*)d_in[5],  *ne_b2 = (const float*)d_in[6];
    const float* ee_w1 = (const float*)d_in[7],  *ee_b1 = (const float*)d_in[8];
    const float* ee_w2 = (const float*)d_in[9],  *ee_b2 = (const float*)d_in[10];
    const float* mp_ew1 = (const float*)d_in[11], *mp_eb1 = (const float*)d_in[12];
    const float* mp_ew2 = (const float*)d_in[13], *mp_eb2 = (const float*)d_in[14];
    const float* mp_nw1 = (const float*)d_in[15], *mp_nb1 = (const float*)d_in[16];
    const float* mp_nw2 = (const float*)d_in[17], *mp_nb2 = (const float*)d_in[18];
    const float* nh_w1 = (const float*)d_in[19], *nh_b1 = (const float*)d_in[20];
    const float* nh_w2 = (const float*)d_in[21], *nh_b2 = (const float*)d_in[22];
    const float* eh_w1 = (const float*)d_in[23], *eh_b1 = (const float*)d_in[24];
    const float* eh_w2 = (const float*)d_in[25], *eh_b2 = (const float*)d_in[26];

    const int N = in_sizes[0] / 6;
    const int E = in_sizes[2] / 8;
    const int L = 4;
    const long long* srcI = ei;
    const long long* dstI = ei + E;

    // ---- workspace carve-up ----
    char* wp = (char*)d_ws;
    auto alloc16 = [&](size_t n) { f16* p = (f16*)wp; wp += ((n * 2 + 255) / 256) * 256; return p; };
    auto allocf  = [&](size_t n) { float* p = (float*)wp; wp += ((n * 4 + 255) / 256) * 256; return p; };

    f16* ne_w1c = alloc16(32 * 96);
    f16* ne_w2c = alloc16(96 * 96);
    f16* ee_w1c = alloc16(32 * 96);
    f16* ee_w2c = alloc16(96 * 96);
    f16* ew1c[L], *ew2c[L], *nw1c[L], *nw2c[L];
    for (int l = 0; l < L; ++l) {
        ew1c[l] = alloc16(288 * 96);
        ew2c[l] = alloc16(96 * 96);
        nw1c[l] = alloc16(192 * 96);
        nw2c[l] = alloc16(96 * 96);
    }
    f16* nh_w1c = alloc16(96 * 96);
    f16* nh_w2c = alloc16(96 * 96);   // [96,1] padded
    f16* eh_w1c = alloc16(288 * 96);
    f16* eh_w2c = alloc16(96 * 96);

    float* h   = allocf((size_t)N * 96);
    float* e   = allocf((size_t)E * 96);
    float* agg = allocf((size_t)N * 96);
    f16*   h16 = alloc16((size_t)N * 96);
    f16*   e16 = alloc16((size_t)E * 96);

    // ---- convert weights to fragment-ordered f16 ----
    auto cvt = [&](const float* s, f16* d, int K, int Nsrc, int Kpad) {
        int total = Kpad * 96;
        cvt_w_frag_kernel<<<dim3((total + 255) / 256), dim3(256), 0, stream>>>(s, d, K, Nsrc, Kpad);
    };
    cvt(ne_w1, ne_w1c, 6, 96, 32);
    cvt(ne_w2, ne_w2c, 96, 96, 96);
    cvt(ee_w1, ee_w1c, 8, 96, 32);
    cvt(ee_w2, ee_w2c, 96, 96, 96);
    for (int l = 0; l < L; ++l) {
        cvt(mp_ew1 + (size_t)l * 288 * 96, ew1c[l], 288, 96, 288);
        cvt(mp_ew2 + (size_t)l * 96 * 96,  ew2c[l], 96, 96, 96);
        cvt(mp_nw1 + (size_t)l * 192 * 96, nw1c[l], 192, 96, 192);
        cvt(mp_nw2 + (size_t)l * 96 * 96,  nw2c[l], 96, 96, 96);
    }
    cvt(nh_w1, nh_w1c, 96, 96, 96);
    cvt(nh_w2, nh_w2c, 96, 1, 96);
    cvt(eh_w1, eh_w1c, 288, 96, 288);
    cvt(eh_w2, eh_w2c, 96, 1, 96);

    const dim3 blk(256);
    const dim3 gN((N + 127) / 128);
    const dim3 gE((E + 127) / 128);

    // ---- encoders (write fp32 master + f16 mirror) ----
    fused_mlp_kernel<<<gN, blk, 0, stream>>>(0, N, 6, x, nullptr, nullptr, nullptr,
                                             nullptr, nullptr,
                                             ne_w1c, ne_b1, ne_w2c, ne_b2,
                                             h, 96, nullptr, 96, h16);
    fused_mlp_kernel<<<gE, blk, 0, stream>>>(0, E, 8, ea, nullptr, nullptr, nullptr,
                                             nullptr, nullptr,
                                             ee_w1c, ee_b1, ee_w2c, ee_b2,
                                             e, 96, nullptr, 96, e16);

    // ---- message-passing layers ----
    for (int l = 0; l < L; ++l) {
        // edge update: e += MLP([h[src], h[dst], e])   (gathers from f16 mirrors)
        fused_mlp_kernel<<<gE, blk, 0, stream>>>(2, E, 288, nullptr, h16, e16, nullptr,
                                                 srcI, dstI,
                                                 ew1c[l], mp_eb1 + l * 96, ew2c[l], mp_eb2 + l * 96,
                                                 e, 96, e, 96, e16);
        // agg = segment_sum(e, dst)  (fp32 atomics)
        hipMemsetAsync(agg, 0, (size_t)N * 96 * sizeof(float), stream);
        seg_sum_kernel<<<dim3((E * 24 + 255) / 256), blk, 0, stream>>>(e, dstI, agg, E);
        // node update: h += MLP([h, agg])
        fused_mlp_kernel<<<gN, blk, 0, stream>>>(1, N, 192, nullptr, h16, nullptr, agg,
                                                 nullptr, nullptr,
                                                 nw1c[l], mp_nb1 + l * 96, nw2c[l], mp_nb2 + l * 96,
                                                 h, 96, h, 96, h16);
    }

    // ---- heads: d_out = [edge_logits(E) | node_logits(N)] ----
    float* out_f = (float*)d_out;
    fused_mlp_kernel<<<gE, blk, 0, stream>>>(2, E, 288, nullptr, h16, e16, nullptr,
                                             srcI, dstI,
                                             eh_w1c, eh_b1, eh_w2c, eh_b2,
                                             out_f, 1, nullptr, 1, nullptr);
    fused_mlp_kernel<<<gN, blk, 0, stream>>>(3, N, 96, nullptr, h16, nullptr, nullptr,
                                             nullptr, nullptr,
                                             nh_w1c, nh_b1, nh_w2c, nh_b2,
                                             out_f + E, 1, nullptr, 1, nullptr);
}
